// TopKSAE_2448131359469
// MI455X (gfx1250) — matrix operand order repaired and checked by reading, not compiled, not run on hardware
//
#include <hip/hip_runtime.h>

// Problem constants (from reference): B=4096, M=768, N=16384, TOPK=32
#define BSZ   4096
#define MDIM  768
#define NDICT 16384
#define TOPK  32

typedef __attribute__((ext_vector_type(16))) __bf16 v16bf;
typedef __attribute__((ext_vector_type(8)))  float  v8f;

union FragBF { v16bf v; uint4 q[2]; };

__device__ inline unsigned short f32_to_bf16(float f) {
    // round-to-nearest-even f32 -> bf16 (bit trick)
    unsigned u = __float_as_uint(f);
    unsigned r = u + 0x7FFFu + ((u >> 16) & 1u);
    return (unsigned short)(r >> 16);
}

// ---------------------------------------------------------------------------
// Kernel 1: S_pre[b,n] = sum_k (X[b,k]-pre_bias[k]) * W[n,k] + latent_bias[n]
// bf16 WMMA, f32 accumulate. Block = 256 threads = 8 waves; block tile 128x128
// (4 row-waves x 2 col-waves); wave tile 32x64 = 2x4 WMMA 16x16x32 fragments.
// K loop over MDIM in steps of 32.
// ---------------------------------------------------------------------------
__global__ __launch_bounds__(256) void sae_encode_gemm(
    const float* __restrict__ X,           // [B, M]
    const float* __restrict__ W,           // [N, M]
    const float* __restrict__ latent_bias, // [N]
    const float* __restrict__ pre_bias,    // [M]
    float* __restrict__ S)                 // [B, N] out (S_pre)
{
    __shared__ __align__(16) unsigned short ax[128][32];  // A tile (rows x K) bf16
    __shared__ __align__(16) unsigned short wt[128][32];  // B tile (cols x K) bf16

    const int tid   = threadIdx.x;
    const int cbase = blockIdx.x * 128;   // N columns
    const int rbase = blockIdx.y * 128;   // batch rows
    const int wave  = tid >> 5;
    const int lane  = tid & 31;
    const int wr    = wave >> 1;          // 0..3  (row dir, 32 rows each)
    const int wc    = wave & 1;           // 0..1  (col dir, 64 cols each)
    const int lm    = lane & 15;
    const int lh    = lane >> 4;

    v8f zero = {};
    v8f acc[2][4];
#pragma unroll
    for (int i = 0; i < 2; ++i)
#pragma unroll
        for (int j = 0; j < 4; ++j) acc[i][j] = zero;

    // Staging: each thread moves 16 elems of X tile and 16 elems of W tile.
    const int sr = tid >> 1;              // 0..127
    const int sc = (tid & 1) * 16;        // 0 or 16

    for (int k0 = 0; k0 < MDIM; k0 += 32) {
        {
            const float* srcx = X + (size_t)(rbase + sr) * MDIM + k0 + sc;
#pragma unroll
            for (int j = 0; j < 16; ++j)
                ax[sr][sc + j] = f32_to_bf16(srcx[j] - pre_bias[k0 + sc + j]);
        }
        {
            const float* srcw = W + (size_t)(cbase + sr) * MDIM + k0 + sc;
#pragma unroll
            for (int j = 0; j < 16; ++j)
                wt[sr][sc + j] = f32_to_bf16(srcw[j]);
            if (k0 + 32 < MDIM)
                __builtin_prefetch(srcw + 32, 0, 0);   // next K slice of W
        }
        __syncthreads();

        // Assemble fragments from LDS per ISA VGPR layouts.
        FragBF a[2], b[4];
#pragma unroll
        for (int t = 0; t < 2; ++t) {
            // A 16x32 bf16: lane half h holds K in {8h..8h+7} U {16+8h..16+8h+7}
            const unsigned short* ap = &ax[wr * 32 + t * 16 + lm][0];
            a[t].q[0] = *(const uint4*)(ap + 8 * lh);
            a[t].q[1] = *(const uint4*)(ap + 16 + 8 * lh);
        }
#pragma unroll
        for (int t = 0; t < 4; ++t) {
            // B 32x16 bf16 (col in lane): lane half h holds K in {8h..8h+7} U {16+8h..16+8h+7}
            const unsigned short* bp = &wt[wc * 64 + t * 16 + lm][0];
            b[t].q[0] = *(const uint4*)(bp + 8 * lh);
            b[t].q[1] = *(const uint4*)(bp + 16 + 8 * lh);
        }

#pragma unroll
        for (int i = 0; i < 2; ++i)
#pragma unroll
            for (int j = 0; j < 4; ++j)
                acc[i][j] = __builtin_amdgcn_wmma_f32_16x16x32_bf16(
                    false, a[i].v, false, b[j].v, (short)0, acc[i][j],
                    false, false);
        __syncthreads();
    }

    // Epilogue: C layout -> lane holds col=lane%16; VGPR v holds row v+8*(lane/16)
#pragma unroll
    for (int i = 0; i < 2; ++i) {
        const int brow0 = rbase + wr * 32 + i * 16 + 8 * lh;
#pragma unroll
        for (int j = 0; j < 4; ++j) {
            const int n  = cbase + wc * 64 + j * 16 + lm;
            const float lb = latent_bias[n];
#pragma unroll
            for (int v = 0; v < 8; ++v)
                S[(size_t)(brow0 + v) * NDICT + n] = acc[i][j][v] + lb;
        }
    }
}

// ---------------------------------------------------------------------------
// Kernel 2: per-row exact top-32 (radix-16 select over monotone-mapped keys),
// ReLU + scatter into S (in place), then sparse decode X_ = S_ @ D + pre_bias.
// One 256-thread block per row; row keys live in 64KB dynamic LDS.
// ---------------------------------------------------------------------------
__device__ inline float inv_key(unsigned k) {
    unsigned u = (k & 0x80000000u) ? (k ^ 0x80000000u) : ~k;
    return __uint_as_float(u);
}

__global__ __launch_bounds__(256) void sae_topk_decode(
    float* __restrict__ S,              // [B, N] in: S_pre, out: S_
    float* __restrict__ Xout,           // [B, M]
    const float* __restrict__ D,        // [N, M]
    const float* __restrict__ pre_bias) // [M]
{
    extern __shared__ unsigned keys[];           // NDICT u32 keys (64 KB)
    __shared__ unsigned s_h2[16][16];            // bank-spread histogram
    __shared__ unsigned s_hist[16];
    __shared__ float    topv[TOPK];
    __shared__ int      topi[TOPK];
    __shared__ int      eqidx[256];
    __shared__ int      s_num, s_eqn;

    const int tid = threadIdx.x;
    const int row = blockIdx.x;
    float* Srow = S + (size_t)row * NDICT;

    // Load row, map f32 -> order-preserving u32
    for (int i = tid; i < NDICT; i += 256) {
        unsigned u = __float_as_uint(Srow[i]);
        keys[i] = (u & 0x80000000u) ? ~u : (u | 0x80000000u);
    }
    if (tid == 0) { s_num = 0; s_eqn = 0; }
    __syncthreads();

    // Radix-16 select of the TOPK-th largest key (exact, 8 passes)
    unsigned prefix = 0;
    int kth = TOPK;
    for (int shift = 28; shift >= 0; shift -= 4) {
        s_h2[tid >> 4][tid & 15] = 0;
        __syncthreads();
        const unsigned hmask = (shift == 28) ? 0u : (~0u << (shift + 4));
        for (int i = tid; i < NDICT; i += 256) {
            unsigned key = keys[i];
            if (((key ^ prefix) & hmask) == 0)
                atomicAdd(&s_h2[(key >> shift) & 15][tid & 15], 1u);
        }
        __syncthreads();
        if (tid < 16) {
            unsigned s = 0;
#pragma unroll
            for (int c = 0; c < 16; ++c) s += s_h2[tid][c];
            s_hist[tid] = s;
        }
        __syncthreads();
        // Uniform bucket choice from the top
        unsigned cum = 0; int chosen = 0;
        for (int bkt = 15; bkt >= 0; --bkt) {
            unsigned c = s_hist[bkt];
            if (cum + c >= (unsigned)kth) { chosen = bkt; break; }
            cum += c;
        }
        kth -= (int)cum;
        prefix |= ((unsigned)chosen) << shift;
        __syncthreads();
    }
    // prefix == key of TOPK-th largest value; kth == #ties to include

    // Collect strictly-greater elements; gather tie indices
    for (int i = tid; i < NDICT; i += 256) {
        unsigned k = keys[i];
        if (k > prefix) {
            int s = atomicAdd(&s_num, 1);
            if (s < TOPK) {
                float f = inv_key(k);
                topv[s] = f > 0.0f ? f : 0.0f;   // ReLU
                topi[s] = i;
            }
        } else if (k == prefix) {
            int e = atomicAdd(&s_eqn, 1);
            if (e < 256) eqidx[e] = i;
        }
    }
    __syncthreads();

    if (tid == 0) {
        int ne = s_eqn; if (ne > 256) ne = 256;
        // smallest tie indices first (matches top_k tie order)
        for (int a = 1; a < ne; ++a) {
            int v = eqidx[a]; int b = a - 1;
            while (b >= 0 && eqidx[b] > v) { eqidx[b + 1] = eqidx[b]; --b; }
            eqidx[b + 1] = v;
        }
        float eqv = inv_key(prefix); eqv = eqv > 0.0f ? eqv : 0.0f;
        int base = s_num; if (base > TOPK) base = TOPK;
        int take = kth; if (take > ne) take = ne;
        if (base + take > TOPK) take = TOPK - base;
        for (int t = 0; t < take; ++t) { topi[base + t] = eqidx[t]; topv[base + t] = eqv; }
        int ns = base + take;
        // sort selected pairs by index -> deterministic FP sum order
        for (int a = 1; a < ns; ++a) {
            int iv = topi[a]; float fv = topv[a]; int b = a - 1;
            while (b >= 0 && topi[b] > iv) {
                topi[b + 1] = topi[b]; topv[b + 1] = topv[b]; --b;
            }
            topi[b + 1] = iv; topv[b + 1] = fv;
        }
        s_num = ns;
    }
    __syncthreads();
    const int nsel = s_num;

    // Write S_: zeros everywhere (coalesced), then scatter the 32 values
    for (int i = tid; i < NDICT; i += 256) Srow[i] = 0.0f;
    __syncthreads();
    if (tid < nsel) Srow[topi[tid]] = topv[tid];

    // Decode: X_[row,:] = sum_t topv[t] * D[topi[t],:] + pre_bias
    for (int m = tid; m < MDIM; m += 256) {
        float acc = pre_bias[m];
        for (int t = 0; t < nsel; ++t)
            acc += topv[t] * D[(size_t)topi[t] * MDIM + m];
        Xout[(size_t)row * MDIM + m] = acc;
    }
}

// ---------------------------------------------------------------------------
extern "C" void kernel_launch(void* const* d_in, const int* in_sizes, int n_in,
                              void* d_out, int out_size, void* d_ws, size_t ws_size,
                              hipStream_t stream) {
    (void)in_sizes; (void)n_in; (void)out_size; (void)d_ws; (void)ws_size;
    const float* X  = (const float*)d_in[0];
    const float* D  = (const float*)d_in[1];
    const float* We = (const float*)d_in[2];
    const float* lb = (const float*)d_in[3];
    const float* pb = (const float*)d_in[4];

    float* S  = (float*)d_out;                              // [B, N]
    float* Xo = (float*)d_out + (size_t)BSZ * NDICT;        // [B, M]

    dim3 grid1(NDICT / 128, BSZ / 128);
    sae_encode_gemm<<<grid1, 256, 0, stream>>>(X, We, lb, pb, S);

    sae_topk_decode<<<BSZ, 256, NDICT * sizeof(unsigned), stream>>>(S, Xo, D, pb);
}
